// VanillaRNN_35442070126759
// MI455X (gfx1250) — matrix operand order repaired and checked by reading, not compiled, not run on hardware
//
#include <hip/hip_runtime.h>
#include <hip/hip_bf16.h>

#define VOCAB  256
#define EMBED  512
#define HIDDEN 1024
#define BATCH  128
#define SEQL   1024

typedef __bf16 bf16;
typedef __attribute__((ext_vector_type(16))) __bf16 v16bf;
typedef __attribute__((ext_vector_type(8)))  __bf16 v8bf;
typedef __attribute__((ext_vector_type(8)))  float  v8f;

// ---------------------------------------------------------------------------
// WMMA fragment loaders (wave32, ISA 7.12.2 layouts, bf16 16x16x32)
// A (16x32, MxK): lane 0-15 => M=lane, K-chunks [k0..k0+7] & [k0+16..k0+23]
//                 lane 16-31 => M=lane-16, chunks shifted by +8
// B (32x16, KxN): lane 0-15 => N=lane, K-chunks [k0..k0+7] & [k0+16..k0+23]
//                 lane 16-31 => N=lane-16, chunks shifted by +8 (same as A)
// Both A rows and B "rows" (weight rows, since B[k][n] = W[n][k]) are
// K-contiguous in memory -> wide aligned loads (2x16B for A and B).
// ---------------------------------------------------------------------------
__device__ __forceinline__ v16bf load_a_frag(const bf16* __restrict__ A, int lda,
                                             int row0, int k0, int lane) {
    int m  = row0 + (lane & 15);
    int kb = k0 + ((lane & 16) ? 8 : 0);
    const bf16* p = A + (size_t)m * lda + kb;
    union { v16bf v; v8bf h[2]; } u;
    u.h[0] = *(const v8bf*)(p);        // K = kb .. kb+7
    u.h[1] = *(const v8bf*)(p + 16);   // K = kb+16 .. kb+23
    return u.v;
}

__device__ __forceinline__ v16bf load_b_frag(const bf16* __restrict__ W, int ldw,
                                             int col0, int k0, int lane) {
    int n  = col0 + (lane & 15);
    int kb = k0 + ((lane & 16) ? 8 : 0);
    const bf16* p = W + (size_t)n * ldw + kb;
    union { v16bf v; v8bf h[2]; } u;
    u.h[0] = *(const v8bf*)(p);        // K = kb .. kb+7
    u.h[1] = *(const v8bf*)(p + 16);   // K = kb+16 .. kb+23
    return u.v;
}

__device__ __forceinline__ v8f wmma_bf16(v16bf a, v16bf b, v8f c) {
    return __builtin_amdgcn_wmma_f32_16x16x32_bf16(false, a, false, b,
                                                   (short)0, c, false, false);
}

// ---------------------------------------------------------------------------
// f32 -> bf16 conversion (weights, embeddings, h0)
// ---------------------------------------------------------------------------
__global__ __launch_bounds__(256) void cvt_bf16_kernel(const float* __restrict__ src,
                                                       bf16* __restrict__ dst, int n) {
    int i = blockIdx.x * 256 + threadIdx.x;
    if (i < n) dst[i] = (bf16)src[i];
}

// ---------------------------------------------------------------------------
// proj[v][h] = emb[v] @ W_ih^T + b_ih[h] + b_hh[h]   (256 x 1024, K=512)
// Collapses the entire [B,S,H] input projection into a 1 MB L2-resident table.
// ---------------------------------------------------------------------------
__global__ __launch_bounds__(256) void proj_gemm_kernel(
    const bf16* __restrict__ embB, const bf16* __restrict__ WihB,
    const float* __restrict__ b_ih, const float* __restrict__ b_hh,
    float* __restrict__ proj)
{
    int lane = threadIdx.x & 31;
    int wave = threadIdx.x >> 5;
    int tile = blockIdx.x * 8 + wave;          // 8 Mgroups x 32 Ngroups = 256
    int ng = tile & 31;
    int mg = tile >> 5;
    int row0 = mg * 32, col0 = ng * 32;
    int nlo = (lane & 15);
    int mbo = (lane & 16) ? 8 : 0;

    v8f acc[2][2];
    #pragma unroll
    for (int ni = 0; ni < 2; ++ni) {
        float bias = b_ih[col0 + ni * 16 + nlo] + b_hh[col0 + ni * 16 + nlo];
        #pragma unroll
        for (int mi = 0; mi < 2; ++mi)
            #pragma unroll
            for (int r = 0; r < 8; ++r) acc[mi][ni][r] = bias;
    }

    #pragma unroll 2
    for (int k = 0; k < EMBED; k += 32) {
        v16bf a0 = load_a_frag(embB, EMBED, row0,      k, lane);
        v16bf a1 = load_a_frag(embB, EMBED, row0 + 16, k, lane);
        v16bf b0 = load_b_frag(WihB, EMBED, col0,      k, lane);
        v16bf b1 = load_b_frag(WihB, EMBED, col0 + 16, k, lane);
        acc[0][0] = wmma_bf16(a0, b0, acc[0][0]);
        acc[0][1] = wmma_bf16(a0, b1, acc[0][1]);
        acc[1][0] = wmma_bf16(a1, b0, acc[1][0]);
        acc[1][1] = wmma_bf16(a1, b1, acc[1][1]);
    }

    #pragma unroll
    for (int mi = 0; mi < 2; ++mi)
        #pragma unroll
        for (int ni = 0; ni < 2; ++ni)
            #pragma unroll
            for (int r = 0; r < 8; ++r)
                proj[(size_t)(row0 + mi * 16 + mbo + r) * HIDDEN
                     + (col0 + ni * 16 + nlo)] = acc[mi][ni][r];
}

// ---------------------------------------------------------------------------
// Device-side helper: output head tile  out[s'] = h @ W_ho^T + b_ho
// (128x256, K=1024; 2x2 tiling; 32 wave-tiles == 4 blocks x 8 waves)
// ---------------------------------------------------------------------------
__device__ __forceinline__ void out_head_tile(
    int sOut, int tile, int lane,
    const bf16* __restrict__ h, const bf16* __restrict__ Who,
    const float* __restrict__ b_ho, float* __restrict__ out)
{
    int ng = tile & 7;                          // 8 Ngroups over V=256
    int mg = tile >> 3;                         // 4 Mgroups over B=128
    int row0 = mg * 32, col0 = ng * 32;
    int nlo = (lane & 15);
    int mbo = (lane & 16) ? 8 : 0;

    v8f acc[2][2];
    #pragma unroll
    for (int ni = 0; ni < 2; ++ni) {
        float bias = b_ho[col0 + ni * 16 + nlo];
        #pragma unroll
        for (int mi = 0; mi < 2; ++mi)
            #pragma unroll
            for (int r = 0; r < 8; ++r) acc[mi][ni][r] = bias;
    }

    #pragma unroll 2
    for (int k = 0; k < HIDDEN; k += 32) {
        v16bf a0 = load_a_frag(h,   HIDDEN, row0,      k, lane);
        v16bf a1 = load_a_frag(h,   HIDDEN, row0 + 16, k, lane);
        v16bf b0 = load_b_frag(Who, HIDDEN, col0,      k, lane);
        v16bf b1 = load_b_frag(Who, HIDDEN, col0 + 16, k, lane);
        acc[0][0] = wmma_bf16(a0, b0, acc[0][0]);
        acc[0][1] = wmma_bf16(a0, b1, acc[0][1]);
        acc[1][0] = wmma_bf16(a1, b0, acc[1][0]);
        acc[1][1] = wmma_bf16(a1, b1, acc[1][1]);
    }

    #pragma unroll
    for (int mi = 0; mi < 2; ++mi)
        #pragma unroll
        for (int ni = 0; ni < 2; ++ni)
            #pragma unroll
            for (int r = 0; r < 8; ++r) {
                int b = row0 + mi * 16 + mbo + r;
                out[((size_t)b * SEQL + sOut) * VOCAB + (col0 + ni * 16 + nlo)]
                    = acc[mi][ni][r];
            }
}

// ---------------------------------------------------------------------------
// Fused per-step kernel.  Key insight: out(s-1) consumes h_s, and hstep(s)
// also consumes only h_s -> they are independent and run concurrently in one
// kernel with zero internal sync, halving the launches on the serial chain.
//   blocks 0..15 : h_{s+1} = tanh(proj[token_s] + h_s @ W_hh^T)   (128x1024)
//   blocks 16..19: out[:,s-1,:] = h_s @ W_ho^T + b_ho             (128x256)
// ---------------------------------------------------------------------------
__global__ __launch_bounds__(256) void rnn_step_fused_kernel(
    int s, int write_f32, const int* __restrict__ seq, const float* __restrict__ proj,
    const bf16* __restrict__ hin, const bf16* __restrict__ Whh,
    const bf16* __restrict__ Who, const float* __restrict__ b_ho,
    float* __restrict__ hF, bf16* __restrict__ hB, float* __restrict__ out)
{
    int lane = threadIdx.x & 31;
    int wave = threadIdx.x >> 5;

    if (blockIdx.x >= 16) {                    // output head for step s-1
        if (s == 0) return;                    // uniform per-block: no out yet
        out_head_tile(s - 1, (blockIdx.x - 16) * 8 + wave, lane, hin, Who, b_ho, out);
        return;
    }

    // ---- h update: 4 Mgroups x 32 Ngroups = 128 wave-tiles, 2x2 tiling ----
    int tile = blockIdx.x * 8 + wave;
    int ng = tile & 31;
    int mg = tile >> 5;
    int row0 = mg * 32, col0 = ng * 32;
    int nlo = (lane & 15);
    int mbo = (lane & 16) ? 8 : 0;

    v8f acc[2][2];
    #pragma unroll
    for (int mi = 0; mi < 2; ++mi)
        #pragma unroll
        for (int r = 0; r < 8; ++r) {
            int tok = seq[(size_t)(row0 + mi * 16 + mbo + r) * SEQL + s];
            const float* pr = proj + (size_t)tok * HIDDEN + col0 + nlo;
            acc[mi][0][r] = pr[0];              // includes b_ih + b_hh
            acc[mi][1][r] = pr[16];
        }

    #pragma unroll 2
    for (int k = 0; k < HIDDEN; k += 32) {
        v16bf a0 = load_a_frag(hin, HIDDEN, row0,      k, lane);
        v16bf a1 = load_a_frag(hin, HIDDEN, row0 + 16, k, lane);
        v16bf b0 = load_b_frag(Whh, HIDDEN, col0,      k, lane);
        v16bf b1 = load_b_frag(Whh, HIDDEN, col0 + 16, k, lane);
        acc[0][0] = wmma_bf16(a0, b0, acc[0][0]);
        acc[0][1] = wmma_bf16(a0, b1, acc[0][1]);
        acc[1][0] = wmma_bf16(a1, b0, acc[1][0]);
        acc[1][1] = wmma_bf16(a1, b1, acc[1][1]);
    }

    #pragma unroll
    for (int mi = 0; mi < 2; ++mi)
        #pragma unroll
        for (int ni = 0; ni < 2; ++ni)
            #pragma unroll
            for (int r = 0; r < 8; ++r) {
                size_t idx = (size_t)(row0 + mi * 16 + mbo + r) * HIDDEN
                             + (col0 + ni * 16 + nlo);
                float v = tanhf(acc[mi][ni][r]);
                hB[idx] = (bf16)v;
                if (write_f32) hF[idx] = v;     // only needed on the last step
            }
}

// Final output head (for s = SEQL-1), standalone.
__global__ __launch_bounds__(256) void rnn_out_kernel(
    int sOut, const bf16* __restrict__ h, const bf16* __restrict__ Who,
    const float* __restrict__ b_ho, float* __restrict__ out)
{
    int lane = threadIdx.x & 31;
    int wave = threadIdx.x >> 5;
    out_head_tile(sOut, blockIdx.x * 8 + wave, lane, h, Who, b_ho, out);
}

// ---------------------------------------------------------------------------
// Host side
// ---------------------------------------------------------------------------
extern "C" void kernel_launch(void* const* d_in, const int* in_sizes, int n_in,
                              void* d_out, int out_size, void* d_ws, size_t ws_size,
                              hipStream_t stream) {
    (void)in_sizes; (void)n_in; (void)out_size; (void)ws_size;

    const int*   seq  = (const int*)  d_in[0];
    const float* h0   = (const float*)d_in[1];
    const float* emb  = (const float*)d_in[2];
    const float* W_ih = (const float*)d_in[3];
    const float* b_ih = (const float*)d_in[4];
    const float* W_hh = (const float*)d_in[5];
    const float* b_hh = (const float*)d_in[6];
    const float* W_ho = (const float*)d_in[7];
    const float* b_ho = (const float*)d_in[8];
    float* out = (float*)d_out;

    // Workspace layout (all offsets 256B-aligned; total ~5.8 MB)
    char* ws = (char*)d_ws;
    size_t off = 0;
    bf16* WhhB = (bf16*)(ws + off); off += (size_t)HIDDEN * HIDDEN * 2;    // 2 MB
    bf16* WhoB = (bf16*)(ws + off); off += (size_t)VOCAB  * HIDDEN * 2;    // 0.5 MB
    bf16* WihB = (bf16*)(ws + off); off += (size_t)HIDDEN * EMBED  * 2;    // 1 MB
    bf16* embB = (bf16*)(ws + off); off += (size_t)VOCAB  * EMBED  * 2;    // 0.25 MB
    bf16* hB   = (bf16*)(ws + off); off += (size_t)2 * BATCH * HIDDEN * 2; // 0.5 MB (ping-pong)
    float* hF  = (float*)(ws + off); off += (size_t)BATCH * HIDDEN * 4;    // 0.5 MB (final h)
    float* proj = (float*)(ws + off); off += (size_t)VOCAB * HIDDEN * 4;   // 1 MB

    const size_t BH = (size_t)BATCH * HIDDEN;

    // 1) Convert weights / embeddings / h0 to bf16.
    auto cvt = [&](const float* s_, bf16* d_, int n) {
        cvt_bf16_kernel<<<(n + 255) / 256, 256, 0, stream>>>(s_, d_, n);
    };
    cvt(W_hh, WhhB, HIDDEN * HIDDEN);
    cvt(W_ho, WhoB, VOCAB * HIDDEN);
    cvt(W_ih, WihB, HIDDEN * EMBED);
    cvt(emb,  embB, VOCAB * EMBED);
    cvt(h0,   hB,   BATCH * HIDDEN);                       // slot 0

    // 2) proj = emb @ W_ih^T + b_ih + b_hh   (folds entire input projection)
    proj_gemm_kernel<<<32, 256, 0, stream>>>(embB, WihB, b_ih, b_hh, proj);

    // 3) Sequential recurrence: ONE fused kernel per step.
    //    Step s: blocks 0-15 compute h_{s+1}; blocks 16-19 emit out(s-1).
    for (int s = 0; s < SEQL; ++s) {
        const bf16* hin  = hB + (size_t)(s & 1)       * BH;
        bf16*       hout = hB + (size_t)((s & 1) ^ 1) * BH;
        int last = (s == SEQL - 1);
        rnn_step_fused_kernel<<<20, 256, 0, stream>>>(s, last, seq, proj, hin,
                                                      WhhB, WhoB, b_ho, hF, hout, out);
    }
    // Trailing output head for the last step (h_SEQL lives in slot SEQL&1 == 0).
    rnn_out_kernel<<<4, 256, 0, stream>>>(SEQL - 1, hB + (size_t)(SEQL & 1) * BH,
                                          WhoB, b_ho, out);

    // 4) h_final -> tail of d_out after the [B,S,V] outputs.
    hipMemcpyAsync(out + (size_t)BATCH * SEQL * VOCAB, hF,
                   BH * sizeof(float), hipMemcpyDeviceToDevice, stream);
}
